// spiking_transformer_dend_12953621364899
// MI455X (gfx1250) — matrix operand order repaired and checked by reading, not compiled, not run on hardware
//
#include <hip/hip_runtime.h>

typedef __attribute__((ext_vector_type(16))) _Float16     v16h;
typedef __attribute__((ext_vector_type(8)))  float        v8f;
typedef __attribute__((ext_vector_type(4)))  unsigned int u32x4;

union AFrag { v16h h; u32x4 q[2]; };
union OFrag { v8f  f; u32x4 q[2]; };

#define BN_EPS 1e-5f

// ---------------------------------------------------------------------------
// Kernel 1: fold BN scale into conv weights, pack f16 PRE-SWIZZLED into the
// exact per-lane WMMA B-fragment order:
//   w1p: [nTile(16)][tap(9)][ks(4)][lane(32)][16 halfs]
//   wrp: [nTile(16)][ks(4)][lane(32)][16 halfs]
// Lane l carries N = l&15, K halfs (l>>4)*8 + (i&7) + (i>>3)*16  (32x16 f16 B layout), so a
// wave's fragment load is one fully coalesced, contiguous 1KB clause.
// ---------------------------------------------------------------------------
__global__ __launch_bounds__(256) void pack_weights(
    const float* __restrict__ w1,
    const float* __restrict__ g1, const float* __restrict__ b1,
    const float* __restrict__ m1, const float* __restrict__ v1,
    const float* __restrict__ wr,
    const float* __restrict__ gr, const float* __restrict__ br,
    const float* __restrict__ mr, const float* __restrict__ vr,
    _Float16* __restrict__ w1p, _Float16* __restrict__ wrp,
    float* __restrict__ bias1, float* __restrict__ biasr)
{
  const int i = blockIdx.x * 256 + threadIdx.x;
  if (i < 294912) {                      // 16*9*4*32*16
    const int ii   = i & 15;
    const int l    = (i >> 4) & 31;
    const int ks   = (i >> 9) & 3;
    const int rest = i >> 11;            // 0..143
    const int tap  = rest % 9;
    const int nT   = rest / 9;
    const int co   = nT * 16 + (l & 15);
    const int ci   = ks * 32 + (l >> 4) * 8 + (ii & 7) + (ii >> 3) * 16;
    const int ky   = tap / 3, kx = tap - ky * 3;
    const float sc = g1[co] * rsqrtf(v1[co] + BN_EPS);
    // input layout w1[co][ci][ky][kx]
    w1p[i] = (_Float16)(w1[((co * 128 + ci) * 3 + ky) * 3 + kx] * sc);
  } else if (i < 294912 + 32768) {       // 16*4*32*16
    const int q  = i - 294912;
    const int ii = q & 15;
    const int l  = (q >> 4) & 31;
    const int ks = (q >> 9) & 3;
    const int nT = q >> 11;
    const int co = nT * 16 + (l & 15);
    const int ci = ks * 32 + (l >> 4) * 8 + (ii & 7) + (ii >> 3) * 16;
    const float sc = gr[co] * rsqrtf(vr[co] + BN_EPS);
    wrp[q] = (_Float16)(wr[co * 128 + ci] * sc);
  }
  if (i < 256) {
    const float sc1 = g1[i] * rsqrtf(v1[i] + BN_EPS);
    bias1[i] = b1[i] - m1[i] * sc1;
    const float scr = gr[i] * rsqrtf(vr[i] + BN_EPS);
    biasr[i] = br[i] - mr[i] * scr;
  }
}

// ---------------------------------------------------------------------------
// Kernel 2: stem conv3x3 (2->128) + BN + LIF over T=4.  K=18 -> plain VALU.
// Spikes stored channels-last f16: s0[t][b][h][w][ci], exact {0,1} values.
// ---------------------------------------------------------------------------
__global__ __launch_bounds__(256) void stem_conv_bn_lif(
    const float* __restrict__ x,  const float* __restrict__ w0,
    const float* __restrict__ g0, const float* __restrict__ b0,
    const float* __restrict__ m0, const float* __restrict__ v0,
    _Float16* __restrict__ s0)
{
  const int gid = blockIdx.x * 256 + threadIdx.x;
  const int co = gid & 127;
  const int w  = (gid >> 7) & 127;
  const int h  = (gid >> 14) & 127;
  const int b  = gid >> 21;
  if (b >= 8) return;

  const float sc = g0[co] * rsqrtf(v0[co] + BN_EPS);
  const float bi = b0[co] - m0[co] * sc;
  float wf[18];
#pragma unroll
  for (int i = 0; i < 18; ++i) wf[i] = w0[co * 18 + i] * sc;

  float vmem = 0.0f;
  for (int t = 0; t < 4; ++t) {
    float acc = bi;
#pragma unroll
    for (int ci = 0; ci < 2; ++ci) {
      const float* xp = x + (size_t)((t * 8 + b) * 2 + ci) * 16384;
#pragma unroll
      for (int ky = 0; ky < 3; ++ky) {
        const int hh = h + ky - 1;
#pragma unroll
        for (int kx = 0; kx < 3; ++kx) {
          const int ww = w + kx - 1;
          float xv = 0.0f;
          if (hh >= 0 && hh < 128 && ww >= 0 && ww < 128) xv = xp[hh * 128 + ww];
          acc = fmaf(xv, wf[ci * 9 + ky * 3 + kx], acc);
        }
      }
    }
    vmem = 0.5f * (vmem + acc);                 // v += (x - v)/tau, tau=2
    const bool sp = (vmem >= 1.0f);             // spike(v - V_TH)
    s0[(size_t)((t * 8 + b) * 16384 + h * 128 + w) * 128 + co] =
        sp ? (_Float16)1.0f : (_Float16)0.0f;
    if (sp) vmem = 0.0f;                        // hard reset
  }
}

// ---------------------------------------------------------------------------
// Kernel 3: fused [conv3x3 + BN + LIF] + [conv1x1 + BN + LIF] + add, WMMA f16.
// Block = 1 batch x (4h x 8w) tile x all 256 co.  8 waves = 2(M) x 4(N).
// B fragments hoisted out of the t loop (weights read ONCE per workgroup);
// all B/A addresses reduced to per-lane base + immediate offsets; output
// written as 32B vector stores (one per accumulator).
// ---------------------------------------------------------------------------
__global__ __launch_bounds__(256) void snn_block_wmma(
    const _Float16* __restrict__ s0,
    const _Float16* __restrict__ w1p,
    const _Float16* __restrict__ wrp,
    const float* __restrict__ bias1,
    const float* __restrict__ biasr,
    float* __restrict__ out)
{
  // [t][pos(6x10 halo)][128 ci + 8 pad] : 65280 B
  __shared__ _Float16 sTile[4 * 60 * 136];

  const int bid = blockIdx.x;
  const int b   = bid >> 9;
  const int tl  = bid & 511;
  const int h0  = (tl >> 4) * 4;
  const int w0  = (tl & 15) * 8;

  // ---- stage spike halo tile (all 4 timesteps) into LDS, zero-padded ----
  for (int vi = threadIdx.x; vi < 3840; vi += 256) {
    const int cg  = vi & 15;        // group of 8 halfs (16B)
    const int pt  = vi >> 4;        // 0..239
    const int pos = pt % 60;
    const int t   = pt / 60;
    const int lh  = pos / 10;
    const int lw  = pos - lh * 10;
    const int h   = h0 - 1 + lh;
    const int w   = w0 - 1 + lw;
    u32x4 val = {0u, 0u, 0u, 0u};
    if (h >= 0 && h < 128 && w >= 0 && w < 128)
      val = *(const u32x4*)(s0 + (size_t)((t * 8 + b) * 16384 + h * 128 + w) * 128 + cg * 8);
    *(u32x4*)&sTile[(t * 60 + pos) * 136 + cg * 8] = val;
  }
  __syncthreads();

  const int lane = threadIdx.x & 31;
  const int wv   = threadIdx.x >> 5;
  const int mRow = wv & 1;        // which 16 spatial rows of the 32
  const int nGrp = wv >> 1;       // 64-channel group
  const int l15  = lane & 15;
  const int lHi  = lane >> 4;
  const int kbL  = lHi * 8;       // A-frag per-lane K base (16-bit A layout)

  // A-matrix row for this lane (lanes 0-15 & 16-31 share M)
  const int mA  = mRow * 16 + l15;
  const int dhA = mA >> 3, dwA = mA & 7;

  // Per-lane weight base pointers: every fragment is base + constant offset.
  const _Float16* bBase  = w1p + (size_t)nGrp * 73728 + (size_t)lane * 16;
  const _Float16* brBase = wrp + (size_t)nGrp * 8192  + (size_t)lane * 16;

  const v8f vzero = {0.f, 0.f, 0.f, 0.f, 0.f, 0.f, 0.f, 0.f};

  v8f accz[4][4];                 // [t][nTile] main-branch accumulators
#pragma unroll
  for (int t = 0; t < 4; ++t)
#pragma unroll
    for (int n = 0; n < 4; ++n) accz[t][n] = vzero;

  // ---- main branch conv3x3: B frags loaded once, reused for all 4 t ----
  for (int tap = 0; tap < 9; ++tap) {
    const int ty  = tap / 3;
    const int tx  = tap - ty * 3;
    const _Float16* aBase = &sTile[((dhA + ty) * 10 + (dwA + tx)) * 136 + kbL];
    const _Float16* bTap  = bBase + tap * 2048;
#pragma unroll
    for (int ks = 0; ks < 4; ++ks) {
      AFrag bb[4];                               // coalesced: lane*32B chunks
#pragma unroll
      for (int n = 0; n < 4; ++n) {
        const _Float16* bp = bTap + n * 18432 + ks * 512;
        bb[n].q[0] = *(const u32x4*)(bp);
        bb[n].q[1] = *(const u32x4*)(bp + 8);
      }
#pragma unroll
      for (int t = 0; t < 4; ++t) {
        AFrag a;                                 // 16-bit A 16x32 layout
        const _Float16* ap = aBase + t * 8160 + ks * 32;
        a.q[0] = *(const u32x4*)(ap);            // K 0-7  (lanes>=16: 8-15)
        a.q[1] = *(const u32x4*)(ap + 16);       // K 16-23(lanes>=16: 24-31)
#pragma unroll
        for (int n = 0; n < 4; ++n)
          accz[t][n] = __builtin_amdgcn_wmma_f32_16x16x32_f16(
              false, a.h, false, bb[n].h, (short)0, accz[t][n], false, false);
      }
    }
  }

  // ---- epilogue: residual conv1x1 per t, then LIF (t-sequential) ----
  float vzs[4][8], vrs[4][8];     // LIF membrane state, both branches
#pragma unroll
  for (int n = 0; n < 4; ++n)
#pragma unroll
    for (int r = 0; r < 8; ++r) { vzs[n][r] = 0.0f; vrs[n][r] = 0.0f; }

  const _Float16* aCtr = &sTile[((dhA + 1) * 10 + (dwA + 1)) * 136 + kbL];
  const int hOut = h0 + mRow * 2 + lHi;          // D rows: m>>3 const over r

  for (int t = 0; t < 4; ++t) {
    v8f accr[4] = {vzero, vzero, vzero, vzero};
#pragma unroll
    for (int ks = 0; ks < 4; ++ks) {
      AFrag a;
      const _Float16* ap = aCtr + t * 8160 + ks * 32;
      a.q[0] = *(const u32x4*)(ap);
      a.q[1] = *(const u32x4*)(ap + 16);
#pragma unroll
      for (int n = 0; n < 4; ++n) {
        const _Float16* bp = brBase + n * 2048 + ks * 512;
        AFrag bb;
        bb.q[0] = *(const u32x4*)(bp);
        bb.q[1] = *(const u32x4*)(bp + 8);
        accr[n] = __builtin_amdgcn_wmma_f32_16x16x32_f16(
            false, a.h, false, bb.h, (short)0, accr[n], false, false);
      }
    }
    // bias + LIF (hard reset, v = (v+x)/2, spike at v>=1), out = sz + sr
#pragma unroll
    for (int n = 0; n < 4; ++n) {
      const int co  = nGrp * 64 + n * 16 + l15;   // D layout: N = lane&15
      const float b1v = bias1[co];
      const float brv = biasr[co];
      OFrag o;
#pragma unroll
      for (int r = 0; r < 8; ++r) {               // D layout: m&7 == r
        const float z = accz[t][n][r] + b1v;
        float Vz = 0.5f * (vzs[n][r] + z);
        const float sz = (Vz >= 1.0f) ? 1.0f : 0.0f;
        vzs[n][r] = (Vz >= 1.0f) ? 0.0f : Vz;
        const float rrv = accr[n][r] + brv;
        float Vr = 0.5f * (vrs[n][r] + rrv);
        const float sr = (Vr >= 1.0f) ? 1.0f : 0.0f;
        vrs[n][r] = (Vr >= 1.0f) ? 0.0f : Vr;
        o.f[r] = sz + sr;
      }
      float* dst = out + (size_t)((t * 8 + b) * 256 + co) * 16384
                       + (size_t)hOut * 128 + w0;
      *(u32x4*)(dst)     = o.q[0];                // 32B contiguous per acc
      *(u32x4*)(dst + 4) = o.q[1];
    }
  }
}

// ---------------------------------------------------------------------------
// Launch: pack weights -> stem+LIF -> fused WMMA block.  Workspace layout:
//   [0, 134217728)          s0 spikes f16  (T*B*H*W*128)
//   [134217728, 134807552)  w1p f16  (swizzled, 294912)
//   [134807552, 134873088)  wrp f16  (swizzled, 32768)
//   [134873088, 134874112)  bias1 f32 (256)
//   [134874112, 134875136)  biasr f32 (256)
// ---------------------------------------------------------------------------
extern "C" void kernel_launch(void* const* d_in, const int* in_sizes, int n_in,
                              void* d_out, int out_size, void* d_ws, size_t ws_size,
                              hipStream_t stream)
{
  const float* x  = (const float*)d_in[0];
  const float* w0 = (const float*)d_in[1];
  const float* g0 = (const float*)d_in[2];
  const float* b0 = (const float*)d_in[3];
  const float* m0 = (const float*)d_in[4];
  const float* v0 = (const float*)d_in[5];
  const float* w1 = (const float*)d_in[6];
  const float* g1 = (const float*)d_in[7];
  const float* b1 = (const float*)d_in[8];
  const float* m1 = (const float*)d_in[9];
  const float* v1 = (const float*)d_in[10];
  const float* wr = (const float*)d_in[11];
  const float* gr = (const float*)d_in[12];
  const float* br = (const float*)d_in[13];
  const float* mr = (const float*)d_in[14];
  const float* vr = (const float*)d_in[15];
  float* out = (float*)d_out;

  char* ws = (char*)d_ws;
  _Float16* s0    = (_Float16*)(ws);
  _Float16* w1p   = (_Float16*)(ws + 134217728u);
  _Float16* wrp   = (_Float16*)(ws + 134807552u);
  float*    bias1 = (float*)(ws + 134873088u);
  float*    biasr = (float*)(ws + 134874112u);

  pack_weights<<<1280, 256, 0, stream>>>(w1, g1, b1, m1, v1,
                                         wr, gr, br, mr, vr,
                                         w1p, wrp, bias1, biasr);
  stem_conv_bn_lif<<<65536, 256, 0, stream>>>(x, w0, g0, b0, m0, v0, s0);
  snn_block_wmma<<<4096, 256, 0, stream>>>(s0, w1p, wrp, bias1, biasr, out);
}
